// WindowAttention_8263517078077
// MI455X (gfx1250) — matrix operand order repaired and checked by reading, not compiled, not run on hardware
//
#include <hip/hip_runtime.h>
#include <hip/hip_bf16.h>

typedef __attribute__((ext_vector_type(16))) _Float16 v16h;
typedef __attribute__((ext_vector_type(8)))  _Float16 v8h;
typedef __attribute__((ext_vector_type(4)))  _Float16 v4h;
typedef __attribute__((ext_vector_type(8)))  float    v8f;
typedef int b128_t __attribute__((__vector_size__(16)));   // matches async-LDS builtin param

union V16H { v16h v; v8h h[2]; };

#define ASG __attribute__((address_space(1)))
#define ASL __attribute__((address_space(3)))

#if defined(__has_builtin)
#if __has_builtin(__builtin_amdgcn_global_load_async_to_lds_b128)
#define HAVE_ASYNC_LDS 1
#endif
#endif
#ifndef HAVE_ASYNC_LDS
#define HAVE_ASYNC_LDS 0
#endif

__device__ __forceinline__ void wait_async0() {
#if __has_builtin(__builtin_amdgcn_s_wait_asynccnt)
    __builtin_amdgcn_s_wait_asynccnt(0);
#elif HAVE_ASYNC_LDS
    asm volatile("s_wait_asynccnt 0x0" ::: "memory");
#endif
}

// ---------------------------------------------------------------------------
// Tiled WMMA GEMM:  C[M,N] = A[M,K] * B[K,N] + bias[N]
// Double-buffered LDS (one barrier per K-stage), register-staged global loads,
// L2 prefetch two stages ahead. 8 wave32s, 128x128 macrotile, BK=32.
// Wave grid 2(m) x 4(n); each wave owns 64x32 = 4x2 WMMA tiles.
// ---------------------------------------------------------------------------
template<typename AT, bool OUT_HALF>
__global__ __launch_bounds__(256)
void gemm_bias_wmma(const AT* __restrict__ A, const float* __restrict__ B,
                    const float* __restrict__ bias, void* __restrict__ Cout,
                    int M, int N, int K)
{
    constexpr int BM = 128, BN = 128, BK = 32;
    __shared__ _Float16 As[2][BM][BK];   // row-major [m][k]
    __shared__ _Float16 Bs[2][BN][BK];   // col-major [n][k] (K contiguous)

    const int t    = threadIdx.x;
    const int lane = t & 31;
    const int wave = t >> 5;       // 0..7
    const int wm   = wave & 1;     // 2 row-waves
    const int wn   = wave >> 1;    // 4 col-waves
    const int l15  = lane & 15;
    const int hi   = lane >> 4;    // 0/1

    const int m0 = blockIdx.y * BM;
    const int n0 = blockIdx.x * BN;

    const v8f vzero = {0.f,0.f,0.f,0.f,0.f,0.f,0.f,0.f};
    v8f acc[4][2];
    for (int i = 0; i < 4; ++i)
        for (int j = 0; j < 2; ++j) acc[i][j] = vzero;

    float4 aR[4];   // staging (f32 A)
    v8h    aRH[2];  // staging (f16 A)
    float4 bR[4];   // staging (f32 B)

    auto loadGlobal = [&](int k0) {
        if constexpr (__is_same(AT, float)) {
            for (int i = 0; i < 4; ++i) {                 // 1024 float4 units
                int idx = t + i * 256;
                int row = idx >> 3, c4 = (idx & 7) << 2;
                aR[i] = *(const float4*)&A[(size_t)(m0 + row) * K + k0 + c4];
            }
        } else {
            for (int i = 0; i < 2; ++i) {                 // 512 v8h units
                int idx = t + i * 256;
                int row = idx >> 2, c8 = (idx & 3) << 3;
                aRH[i] = *(const v8h*)&A[(size_t)(m0 + row) * K + k0 + c8];
            }
        }
        for (int i = 0; i < 4; ++i) {                     // 1024 float4 units
            int idx = t + i * 256;
            int kr = idx >> 5, c4 = (idx & 31) << 2;
            bR[i] = *(const float4*)&B[(size_t)(k0 + kr) * N + n0 + c4];
        }
    };

    auto prefetchStage = [&](int k0) {                    // prime L2 (HBM latency)
        if constexpr (__is_same(AT, float)) {
            for (int i = 0; i < 4; ++i) {
                int idx = t + i * 256;
                int row = idx >> 3, c4 = (idx & 7) << 2;
                __builtin_prefetch(&A[(size_t)(m0 + row) * K + k0 + c4], 0, 1);
            }
        } else {
            for (int i = 0; i < 2; ++i) {
                int idx = t + i * 256;
                int row = idx >> 2, c8 = (idx & 3) << 3;
                __builtin_prefetch(&A[(size_t)(m0 + row) * K + k0 + c8], 0, 1);
            }
        }
        for (int i = 0; i < 4; ++i) {
            int idx = t + i * 256;
            int kr = idx >> 5, c4 = (idx & 31) << 2;
            __builtin_prefetch(&B[(size_t)(k0 + kr) * N + n0 + c4], 0, 1);
        }
    };

    auto storeLds = [&](int buf) {
        if constexpr (__is_same(AT, float)) {
            for (int i = 0; i < 4; ++i) {
                int idx = t + i * 256;
                int row = idx >> 3, c4 = (idx & 7) << 2;
                v4h hh = { (_Float16)aR[i].x, (_Float16)aR[i].y,
                           (_Float16)aR[i].z, (_Float16)aR[i].w };
                *(v4h*)&As[buf][row][c4] = hh;
            }
        } else {
            for (int i = 0; i < 2; ++i) {
                int idx = t + i * 256;
                int row = idx >> 2, c8 = (idx & 3) << 3;
                *(v8h*)&As[buf][row][c8] = aRH[i];
            }
        }
        for (int i = 0; i < 4; ++i) {
            int idx = t + i * 256;
            int kr = idx >> 5, c4 = (idx & 31) << 2;
            Bs[buf][c4 + 0][kr] = (_Float16)bR[i].x;
            Bs[buf][c4 + 1][kr] = (_Float16)bR[i].y;
            Bs[buf][c4 + 2][kr] = (_Float16)bR[i].z;
            Bs[buf][c4 + 3][kr] = (_Float16)bR[i].w;
        }
    };

    auto compute = [&](int buf) {
        V16H a[4], bf[2];
        const int koff = hi * 8;
        for (int mt = 0; mt < 4; ++mt) {
            int r = wm * 64 + mt * 16 + l15;
            a[mt].h[0] = *(const v8h*)&As[buf][r][koff];
            a[mt].h[1] = *(const v8h*)&As[buf][r][16 + koff];
        }
        for (int nt = 0; nt < 2; ++nt) {
            int c = wn * 32 + nt * 16 + l15;
            bf[nt].h[0] = *(const v8h*)&Bs[buf][c][hi * 8];
            bf[nt].h[1] = *(const v8h*)&Bs[buf][c][16 + hi * 8];
        }
        for (int mt = 0; mt < 4; ++mt)
            for (int nt = 0; nt < 2; ++nt)
                acc[mt][nt] = __builtin_amdgcn_wmma_f32_16x16x32_f16(
                    false, a[mt].v, false, bf[nt].v, (short)0, acc[mt][nt], false, false);
    };

    const int S = K / BK;          // 16 stages
    loadGlobal(0);
    storeLds(0);
    __syncthreads();
    for (int s = 0; s < S; ++s) {
        if (s + 1 < S) {           // fill other buffer while computing this one
            loadGlobal((s + 1) * BK);
            storeLds((s + 1) & 1);
        }
        if (s + 2 < S) prefetchStage((s + 2) * BK);
        compute(s & 1);
        __syncthreads();           // single barrier per stage (disjoint buffers)
    }

    // ---- epilogue: + bias, store ----
    for (int mt = 0; mt < 4; ++mt)
        for (int nt = 0; nt < 2; ++nt)
            for (int r = 0; r < 8; ++r) {
                int row = m0 + wm * 64 + mt * 16 + r + hi * 8;
                int col = n0 + wn * 32 + nt * 16 + l15;
                float v = acc[mt][nt][r] + bias[col];
                if constexpr (OUT_HALF)
                    ((_Float16*)Cout)[(size_t)row * N + col] = (_Float16)v;
                else
                    ((float*)Cout)[(size_t)row * N + col] = v;
            }
}

// ---------------------------------------------------------------------------
// Fused window attention: one block (4 wave32s) per (window, head).
// N=49 padded to 64, d=32. S=QK^T (4 WMMA/wave), bias+softmax, P*V (4 WMMA/wave).
// Q/K staged to LDS with async global->LDS copies (ASYNCcnt) when available.
// ---------------------------------------------------------------------------
__global__ __launch_bounds__(128)
void win_attn(const _Float16* __restrict__ qkv, const float* __restrict__ bias_table,
              _Float16* __restrict__ aout)
{
    constexpr int N = 49, NP = 64, D = 32, C3 = 1536, C = 512;
    __shared__ _Float16 Qs[NP][D];
    __shared__ _Float16 Ks[NP][D];
    __shared__ _Float16 Vt[D][NP];     // V transposed: [e][n]
    __shared__ _Float16 Ps[NP][NP];    // softmax(P) in f16
    __shared__ float    Bt[13 * 13 * 16];

    const int t    = threadIdx.x;
    const int lane = t & 31;
    const int w    = t >> 5;          // wave 0..3 -> row strip w*16
    const int l15  = lane & 15;
    const int hi   = lane >> 4;

    const int b = blockIdx.x >> 4;    // window
    const int h = blockIdx.x & 15;    // head

    // relative-position bias table -> LDS (hits L2 across 32768 blocks)
    for (int i = t; i < 13 * 13 * 16; i += 128) Bt[i] = bias_table[i];

    const v8h hzero = {0,0,0,0,0,0,0,0};

    // phase 0: zero the pad rows of Q/K and all of Vt (disjoint from async writes)
    for (int u = t; u < NP * 4; u += 128) {
        int row = u >> 2, c8 = (u & 3) << 3;
        if (row >= N) {
            *(v8h*)&Qs[row][c8] = hzero;
            *(v8h*)&Ks[row][c8] = hzero;
        }
    }
    for (int u = t; u < (D * NP) / 8; u += 128)   // 256 v8h chunks
        ((v8h*)Vt)[u] = hzero;
    __syncthreads();

    // phase 1: stage Q,K (async when available) and V (transposed) for rows < N
    for (int u = t; u < NP * 4; u += 128) {       // 64 rows x 4 chunks of 8 halfs
        int row = u >> 2, c8 = (u & 3) << 3;
        if (row < N) {
            const size_t base = ((size_t)b * N + row) * C3 + h * D + c8;
#if HAVE_ASYNC_LDS
            __builtin_amdgcn_global_load_async_to_lds_b128(
                (ASG b128_t*)&qkv[base],     (ASL b128_t*)&Qs[row][c8], 0, 0);
            __builtin_amdgcn_global_load_async_to_lds_b128(
                (ASG b128_t*)&qkv[base + C], (ASL b128_t*)&Ks[row][c8], 0, 0);
#else
            *(v8h*)&Qs[row][c8] = *(const v8h*)&qkv[base];
            *(v8h*)&Ks[row][c8] = *(const v8h*)&qkv[base + C];
#endif
            v8h v = *(const v8h*)&qkv[base + 2 * C];
            for (int j = 0; j < 8; ++j) Vt[c8 + j][row] = v[j];
        }
    }
    wait_async0();
    __syncthreads();

    // ---- S = Q * K^T for this wave's 16-row strip ----
    V16H aq;
    {
        int r = w * 16 + l15;
        int koff = hi * 8;
        aq.h[0] = *(const v8h*)&Qs[r][koff];
        aq.h[1] = *(const v8h*)&Qs[r][16 + koff];
    }
    const v8f vzero = {0.f,0.f,0.f,0.f,0.f,0.f,0.f,0.f};
    v8f s[4];
    for (int nt = 0; nt < 4; ++nt) {
        V16H bk;
        int c = nt * 16 + l15;
        bk.h[0] = *(const v8h*)&Ks[c][hi * 8];
        bk.h[1] = *(const v8h*)&Ks[c][16 + hi * 8];
        s[nt] = __builtin_amdgcn_wmma_f32_16x16x32_f16(
            false, aq.v, false, bk.v, (short)0, vzero, false, false);
    }

    // ---- scale + relative position bias + padding mask (branchless) ----
    const float scale = 0.17677669529663687f;   // 1/sqrt(32)
    int col4[4], rj4[4], cj4[4];
    for (int nt = 0; nt < 4; ++nt) {
        int col  = nt * 16 + l15;
        int colc = (col < N) ? col : 0;         // clamp for safe LDS index
        col4[nt] = col;
        rj4[nt]  = colc / 7;
        cj4[nt]  = colc - rj4[nt] * 7;
    }
    for (int r = 0; r < 8; ++r) {
        int row  = w * 16 + r + hi * 8;
        int rowc = (row < N) ? row : 0;         // rows >= N only need finite values
        int ri = rowc / 7, ci = rowc - ri * 7;
        for (int nt = 0; nt < 4; ++nt) {
            int idx = (ri - rj4[nt] + 6) * 13 + (ci - cj4[nt] + 6);
            float bval = Bt[idx * 16 + h];
            float sv = s[nt][r] * scale + bval;
            s[nt][r] = (col4[nt] < N) ? sv : -1e30f;
        }
    }

    // ---- softmax per row; C-layout: row r+(hi*8), col l15 per 16-lane group ----
    for (int r = 0; r < 8; ++r) {
        float m = fmaxf(fmaxf(s[0][r], s[1][r]), fmaxf(s[2][r], s[3][r]));
        for (int off = 8; off >= 1; off >>= 1) m = fmaxf(m, __shfl_xor(m, off, 32));
        float p[4], sum = 0.f;
        for (int nt = 0; nt < 4; ++nt) { p[nt] = __expf(s[nt][r] - m); sum += p[nt]; }
        for (int off = 8; off >= 1; off >>= 1) sum += __shfl_xor(sum, off, 32);
        float inv = 1.f / sum;
        int row = w * 16 + r + hi * 8;
        for (int nt = 0; nt < 4; ++nt)
            Ps[row][nt * 16 + l15] = (_Float16)(p[nt] * inv);
    }
    __syncthreads();

    // ---- O = P * V  (K=64 -> two 32-chunks; output 16x32 per wave) ----
    v8f o[2] = { vzero, vzero };
    for (int kt = 0; kt < 2; ++kt) {
        V16H ap;
        int r = w * 16 + l15;
        int kof = kt * 32 + hi * 8;
        ap.h[0] = *(const v8h*)&Ps[r][kof];
        ap.h[1] = *(const v8h*)&Ps[r][kof + 16];
        for (int nt = 0; nt < 2; ++nt) {
            V16H bv;
            int c = nt * 16 + l15;
            bv.h[0] = *(const v8h*)&Vt[c][kt * 32 + hi * 8];
            bv.h[1] = *(const v8h*)&Vt[c][kt * 32 + 16 + hi * 8];
            o[nt] = __builtin_amdgcn_wmma_f32_16x16x32_f16(
                false, ap.v, false, bv.v, (short)0, o[nt], false, false);
        }
    }

    // ---- store [b, n, h*32+e] as f16 for the projection GEMM ----
    for (int nt = 0; nt < 2; ++nt)
        for (int r = 0; r < 8; ++r) {
            int row = w * 16 + r + hi * 8;
            if (row < N) {
                int e = nt * 16 + l15;
                aout[((size_t)b * N + row) * C + h * D + e] = (_Float16)o[nt][r];
            }
        }
}

// ---------------------------------------------------------------------------
extern "C" void kernel_launch(void* const* d_in, const int* in_sizes, int n_in,
                              void* d_out, int out_size, void* d_ws, size_t ws_size,
                              hipStream_t stream) {
    const float* x          = (const float*)d_in[0];   // [2048,49,512]
    const float* w_qkv      = (const float*)d_in[1];   // [512,1536]
    const float* b_qkv      = (const float*)d_in[2];   // [1536]
    const float* bias_table = (const float*)d_in[3];   // [13,13,16]
    const float* w_proj     = (const float*)d_in[4];   // [512,512]
    const float* b_proj     = (const float*)d_in[5];   // [512]
    float*       out        = (float*)d_out;           // [2048,49,512]

    const int M = 2048 * 49;                           // 100352 (multiple of 128)
    _Float16* qkv  = (_Float16*)d_ws;                  // [M,1536] f16
    _Float16* aout = qkv + (size_t)M * 1536;           // [M, 512] f16

    dim3 blk(256);
    // 1) QKV projection: fp32 in -> f16 qkv out
    gemm_bias_wmma<float, true><<<dim3(1536 / 128, M / 128), blk, 0, stream>>>(
        x, w_qkv, b_qkv, (void*)qkv, M, 1536, 512);

    // 2) fused window attention: one block per (window, head)
    win_attn<<<dim3(2048 * 16), dim3(128), 0, stream>>>(qkv, bias_table, aout);

    // 3) output projection: f16 in -> fp32 out (+bias)
    gemm_bias_wmma<_Float16, false><<<dim3(512 / 128, M / 128), blk, 0, stream>>>(
        aout, w_proj, b_proj, (void*)out, M, 512, 512);
}
